// OrigMLPBlock_2619930051312
// MI455X (gfx1250) — matrix operand order repaired and checked by reading, not compiled, not run on hardware
//
#include <hip/hip_runtime.h>
#include <hip/hip_bf16.h>
#include <math.h>

#define T_TOK 1024
#define HDIM  1024
#define IDIM  1024
#define NEXP  8
#define TOPK  2
#define TWO_I 2048
#define EPS   1e-5f
#define ALPHA 1.702f
#define LIMIT 7.0f

// LDS A-tile layout: each 512-element (1024B) chunk is followed by 16B pad.
// Chunk stride = 520 elements = 1040B: 16B-aligned (ds_load_b128) and
// 1040/4 % 64 == 4 -> 4-bank skew per chunk => conflict-free frag reads.
// This layout is exactly what the TDM produces with pad_interval=7 (1024B)
// and pad_amount=3 (4 DWORDs = 16B).
#define LDSA_CHUNK 520
#define LDSA_ELEMS (32 * 2 * LDSA_CHUNK)   // 32 rows x 2 chunks

typedef __attribute__((ext_vector_type(16))) __bf16 v16bf;
typedef __attribute__((ext_vector_type(8)))  __bf16 v8bf;
typedef __attribute__((ext_vector_type(8)))  float  v8f;

#if defined(__gfx1250__) && __has_builtin(__builtin_amdgcn_tensor_load_to_lds)
#define USE_TDM 1
typedef unsigned int u32x4 __attribute__((ext_vector_type(4)));
typedef int          i32x4 __attribute__((ext_vector_type(4)));
typedef int          i32x8 __attribute__((ext_vector_type(8)));
#endif

__device__ __forceinline__ int ldsa_off(int r, int k) {
  return (2 * r + (k >> 9)) * LDSA_CHUNK + (k & 511);
}

// ---------------------------------------------------------------------------
// Fragment loaders (WMMA 16x16x32 bf16; contiguous k order k = 16h + s used by this code for both A and B)
// A (16x32, MxK): lane m=l&15; lanes 0-15 hold K=0..15
//                 lanes 16-31 hold K=16..31
// B (32x16, KxN): lane n=l&15; lanes 0-15 hold K=0..15, lanes 16-31 K=16..31
// ---------------------------------------------------------------------------
__device__ __forceinline__ v16bf load_a_frag(const __bf16* sA, int mBase,
                                             int k0, int lane) {
  const int m  = mBase + (lane & 15);
  const int kh = (lane >> 4) << 4;             // 0 or 16
  const __bf16* p = sA + ldsa_off(m, k0);
  v8bf lo = *reinterpret_cast<const v8bf*>(p + kh);        // K = kh..kh+7
  v8bf hi = *reinterpret_cast<const v8bf*>(p + kh + 8);    // K = kh+8..kh+15
  v16bf a;
#pragma unroll
  for (int i = 0; i < 8; ++i) { a[i] = lo[i]; a[i + 8] = hi[i]; }
  return a;
}

__device__ __forceinline__ v16bf load_b_frag(const __hip_bfloat16* Wbase,
                                             int ldk, int n0, int k0, int lane) {
  const int n  = lane & 15;
  const int kb = k0 + ((lane >> 4) << 4);      // 16 contiguous K per lane
  return *reinterpret_cast<const v16bf*>(Wbase + (size_t)(n0 + n) * ldk + kb);
}

// ---------------------------------------------------------------------------
__global__ void k_zero(int* counts) {
  if (threadIdx.x < NEXP) counts[threadIdx.x] = 0;
}

// ---------------------------------------------------------------------------
// RMSNorm + gate + top-2 softmax + expert counting. 1 block / token.
// ---------------------------------------------------------------------------
__global__ void __launch_bounds__(128)
k_route(const __hip_bfloat16* __restrict__ x,
        const float*          __restrict__ nscale,
        const __hip_bfloat16* __restrict__ gw,
        const __hip_bfloat16* __restrict__ gb,
        __hip_bfloat16*       __restrict__ t_norm,
        int*                  __restrict__ idx_tk,
        float*                __restrict__ ew_tk,
        int*                  __restrict__ counts) {
  const int t   = blockIdx.x;
  const int tid = threadIdx.x;
  __shared__ float red[128];
  __shared__ float glog[NEXP];

  const __hip_bfloat16* xr = x + (size_t)t * HDIM;
  float xv[8], tf[8];
  float ss = 0.f;
#pragma unroll
  for (int j = 0; j < 8; ++j) {
    float v = __bfloat162float(xr[tid * 8 + j]);
    xv[j] = v; ss += v * v;
  }
  red[tid] = ss; __syncthreads();
  for (int s = 64; s > 0; s >>= 1) { if (tid < s) red[tid] += red[tid + s]; __syncthreads(); }
  const float scale = rsqrtf(red[0] / (float)HDIM + EPS);
  __syncthreads();

#pragma unroll
  for (int j = 0; j < 8; ++j) {
    float v = xv[j] * scale * nscale[tid * 8 + j];
    __hip_bfloat16 b = __float2bfloat16(v);
    tf[j] = __bfloat162float(b);
    t_norm[(size_t)t * HDIM + tid * 8 + j] = b;
  }

  for (int e = 0; e < NEXP; ++e) {
    const __hip_bfloat16* gr = gw + (size_t)e * HDIM + tid * 8;
    float p = 0.f;
#pragma unroll
    for (int j = 0; j < 8; ++j) p += tf[j] * __bfloat162float(gr[j]);
    red[tid] = p; __syncthreads();
    for (int s = 64; s > 0; s >>= 1) { if (tid < s) red[tid] += red[tid + s]; __syncthreads(); }
    if (tid == 0) glog[e] = red[0] + __bfloat162float(gb[e]);
    __syncthreads();
  }

  if (tid == 0) {
    float v0 = -1e30f, v1 = -1e30f; int i0 = 0, i1 = 0;
    for (int e = 0; e < NEXP; ++e) { float v = glog[e]; if (v > v0) { v0 = v; i0 = e; } }
    for (int e = 0; e < NEXP; ++e) { if (e == i0) continue; float v = glog[e]; if (v > v1) { v1 = v; i1 = e; } }
    float b  = __expf(v1 - v0);
    float w0 = 1.0f / (1.0f + b);
    float w1 = b * w0;
    w0 = __bfloat162float(__float2bfloat16(w0));
    w1 = __bfloat162float(__float2bfloat16(w1));
    idx_tk[t * 2 + 0] = i0; idx_tk[t * 2 + 1] = i1;
    ew_tk[t * 2 + 0] = w0;  ew_tk[t * 2 + 1] = w1;
    atomicAdd(&counts[i0], 1);
    atomicAdd(&counts[i1], 1);
  }
}

// ---------------------------------------------------------------------------
__global__ void k_prefix(const int* __restrict__ counts,
                         int* __restrict__ offs, int* __restrict__ cursor) {
  if (threadIdx.x == 0 && blockIdx.x == 0) {
    int off = 0;
    for (int e = 0; e < NEXP; ++e) { offs[e] = off; cursor[e] = off; off += counts[e]; }
  }
}

// ---------------------------------------------------------------------------
__global__ void k_scatter(const int* __restrict__ idx_tk,
                          const float* __restrict__ ew_tk,
                          int* __restrict__ cursor,
                          int* __restrict__ perm,
                          int* __restrict__ inv,
                          float* __restrict__ slot_w) {
  const int t = blockIdx.x * blockDim.x + threadIdx.x;
  if (t >= T_TOK) return;
#pragma unroll
  for (int k = 0; k < TOPK; ++k) {
    int e   = idx_tk[t * 2 + k];
    int pos = atomicAdd(&cursor[e], 1);
    perm[pos]       = t;
    inv[t * 2 + k]  = pos;
    slot_w[pos]     = ew_tk[t * 2 + k];
  }
}

// ---------------------------------------------------------------------------
// Grouped up-projection + fused SwiGLU.
// Block = 256 thr (8 waves). Tile: 32 slots x 256 up-channels; K = H = 1024.
// Each wave: 2 A frags x 2 B frags -> 4 independent WMMAs/step; B fragments
// are software-pipelined one K-step ahead into dedicated registers.
// grid = (TWO_I/256, T/32, NEXP)
// ---------------------------------------------------------------------------
__global__ void __launch_bounds__(256)
k_up(const __hip_bfloat16* __restrict__ t_norm,
     const __hip_bfloat16* __restrict__ mlp1_w,
     const __hip_bfloat16* __restrict__ mlp1_b,
     const int*  __restrict__ perm,
     const int*  __restrict__ offs,
     const int*  __restrict__ counts,
     __hip_bfloat16* __restrict__ h_act) {
  const int e   = blockIdx.z;
  const int cnt = counts[e];
  const int m0  = blockIdx.y * 32;
  if (m0 >= cnt) return;
  const int seg  = offs[e];
  const int tid  = threadIdx.x;
  const int wave = tid >> 5, lane = tid & 31;
  const int n0   = blockIdx.x * 256 + wave * 32;

  __shared__ __align__(16) __bf16 sA[LDSA_ELEMS];
  __shared__ int sTok[32];
  if (tid < 32) {
    int rr = m0 + tid; if (rr >= cnt) rr = cnt - 1;
    sTok[tid] = perm[seg + rr];
  }
  __syncthreads();

  // stage full A tile: 32 rows x 1024 bf16, 16B chunks, coalesced
#pragma unroll
  for (int i = 0; i < 16; ++i) {
    const int c   = tid + i * 256;
    const int r   = c >> 7;                     // 128 chunks per row
    const int col = (c & 127) << 3;
    const uint4 v = *reinterpret_cast<const uint4*>(
        t_norm + (size_t)sTok[r] * HDIM + col);
    *reinterpret_cast<uint4*>(sA + ldsa_off(r, col)) = v;
  }
  __syncthreads();

  const __hip_bfloat16* Wb = mlp1_w + (size_t)e * TWO_I * HDIM;
  v8f c00 = {}, c01 = {}, c10 = {}, c11 = {};
  v16bf b0 = load_b_frag(Wb, HDIM, n0,      0, lane);
  v16bf b1 = load_b_frag(Wb, HDIM, n0 + 16, 0, lane);
  for (int k0 = 0; k0 < HDIM - 32; k0 += 32) {
    v16bf a0  = load_a_frag(sA, 0,  k0, lane);
    v16bf a1  = load_a_frag(sA, 16, k0, lane);
    v16bf nb0 = load_b_frag(Wb, HDIM, n0,      k0 + 32, lane);
    v16bf nb1 = load_b_frag(Wb, HDIM, n0 + 16, k0 + 32, lane);
    if (k0 + 256 < HDIM) {
      const int n = lane & 15, kb = k0 + 256 + ((lane >> 4) << 4);
      __builtin_prefetch(Wb + (size_t)(n0 + n) * HDIM + kb, 0, 3);
    }
    c00 = __builtin_amdgcn_wmma_f32_16x16x32_bf16(false, a0, false, b0, (short)0, c00, false, false);
    c10 = __builtin_amdgcn_wmma_f32_16x16x32_bf16(false, a1, false, b0, (short)0, c10, false, false);
    c01 = __builtin_amdgcn_wmma_f32_16x16x32_bf16(false, a0, false, b1, (short)0, c01, false, false);
    c11 = __builtin_amdgcn_wmma_f32_16x16x32_bf16(false, a1, false, b1, (short)0, c11, false, false);
    b0 = nb0; b1 = nb1;
  }
  {
    const int k0 = HDIM - 32;
    v16bf a0 = load_a_frag(sA, 0,  k0, lane);
    v16bf a1 = load_a_frag(sA, 16, k0, lane);
    c00 = __builtin_amdgcn_wmma_f32_16x16x32_bf16(false, a0, false, b0, (short)0, c00, false, false);
    c10 = __builtin_amdgcn_wmma_f32_16x16x32_bf16(false, a1, false, b0, (short)0, c10, false, false);
    c01 = __builtin_amdgcn_wmma_f32_16x16x32_bf16(false, a0, false, b1, (short)0, c01, false, false);
    c11 = __builtin_amdgcn_wmma_f32_16x16x32_bf16(false, a1, false, b1, (short)0, c11, false, false);
  }

  // Epilogue: bias + SwiGLU on interleaved (gate, linear) channel pairs.
  const int n  = lane & 15;
  const int mh = (lane >> 4) << 3;
#pragma unroll
  for (int nh = 0; nh < 2; ++nh) {
    const int nc = n0 + nh * 16 + n;
    const float bias = __bfloat162float(mlp1_b[(size_t)e * TWO_I + nc]);
#pragma unroll
    for (int half = 0; half < 2; ++half) {
      v8f c = nh ? (half ? c11 : c01) : (half ? c10 : c00);
#pragma unroll
      for (int r = 0; r < 8; ++r) {
        float val   = c[r] + bias;
        float other = __shfl_xor(val, 1, 32);
        const int m = half * 16 + r + mh;
        if (((n & 1) == 0) && (m0 + m < cnt)) {
          float g = fminf(val, LIMIT);
          float l = fminf(fmaxf(other, -LIMIT), LIMIT);
          float s = 1.0f / (1.0f + __expf(-ALPHA * g));
          float act = g * s * (l + 1.0f);
          const int slot = seg + m0 + m;
          h_act[(size_t)slot * IDIM + (nc >> 1)] = __float2bfloat16(act);
        }
      }
    }
  }
}

// ---------------------------------------------------------------------------
// Grouped down-projection: 32 slots x 256 hidden-channels; K = I = 1024.
// A tile staged via Tensor Data Mover (TDM) when available: one
// tensor_load_to_lds moves the 32x1024 bf16 tile (with hardware zero-fill
// for OOB rows and LDS padding matching our bank-skewed layout), then
// s_wait_tensorcnt + barrier. grid = (HDIM/256, T/32, NEXP)
// ---------------------------------------------------------------------------
__global__ void __launch_bounds__(256)
k_down(const __hip_bfloat16* __restrict__ h_act,
       const __hip_bfloat16* __restrict__ mlp2_w,
       const __hip_bfloat16* __restrict__ mlp2_b,
       const int*   __restrict__ offs,
       const int*   __restrict__ counts,
       const float* __restrict__ slot_w,
       float*       __restrict__ down_buf) {
  const int e   = blockIdx.z;
  const int cnt = counts[e];
  const int m0  = blockIdx.y * 32;
  if (m0 >= cnt) return;
  const int seg  = offs[e];
  const int tid  = threadIdx.x;
  const int wave = tid >> 5, lane = tid & 31;
  const int n0   = blockIdx.x * 256 + wave * 32;

  __shared__ __align__(16) __bf16 sA[LDSA_ELEMS];

#if defined(USE_TDM)
  if (wave == 0) {
    const unsigned long long gaddr =
        (unsigned long long)(const void*)(h_act + (size_t)(seg + m0) * IDIM);
    const unsigned int ldsaddr = (unsigned int)(size_t)(&sA[0]);
    const int rows = cnt - m0;                  // OOB rows read as zero
    // D# group0 (ISA 8.3): count=1 | lds_addr | global_addr | type=2
    u32x4 g0;
    g0[0] = 1u;                                  // count=1, user mode
    g0[1] = ldsaddr;                             // lds_addr (bytes)
    g0[2] = (unsigned int)(gaddr & 0xFFFFFFFFull);
    g0[3] = (unsigned int)((gaddr >> 32) & 0x1FFFFFFull) | (2u << 30); // type=2
    // D# group1 (ISA 8.4)
    i32x8 g1;
    g1[0] = (1 << 16)      // data_size = 2 bytes
          | (1 << 20)      // pad_enable
          | (7 << 22)      // pad_interval: 256 DWORDs = 1024B
          | (3 << 25);     // pad_amount: 4 DWORDs = 16B
    g1[1] = (int)((1024u & 0xFFFFu) << 16);      // tensor_dim0 = 1024
    g1[2] = (int)(((unsigned)rows & 0xFFFFu) << 16); // tensor_dim1 = rows
    g1[3] = (int)(1024u << 16);                  // tile_dim0 = 1024
    g1[4] = 32;                                  // tile_dim1 = 32, tile_dim2 = 0
    g1[5] = 1024;                                // tensor_dim0_stride = 1024
    g1[6] = 0;
    g1[7] = 0;
    i32x4 gz4 = {0, 0, 0, 0};
    i32x8 gz8 = {0, 0, 0, 0, 0, 0, 0, 0};
    __builtin_amdgcn_tensor_load_to_lds(g0, g1, gz4, gz4, gz8, 0);
    __builtin_amdgcn_s_wait_tensorcnt(0);
  }
  __syncthreads();
#else
  // fallback: manual coalesced staging
#pragma unroll
  for (int i = 0; i < 16; ++i) {
    const int c   = tid + i * 256;
    int r         = c >> 7;
    const int col = (c & 127) << 3;
    int rr = m0 + r; if (rr >= cnt) rr = cnt - 1;
    const uint4 v = *reinterpret_cast<const uint4*>(
        h_act + (size_t)(seg + rr) * IDIM + col);
    *reinterpret_cast<uint4*>(sA + ldsa_off(r, col)) = v;
  }
  __syncthreads();
#endif

  const __hip_bfloat16* Wb = mlp2_w + (size_t)e * HDIM * IDIM;
  v8f c00 = {}, c01 = {}, c10 = {}, c11 = {};
  v16bf b0 = load_b_frag(Wb, IDIM, n0,      0, lane);
  v16bf b1 = load_b_frag(Wb, IDIM, n0 + 16, 0, lane);
  for (int k0 = 0; k0 < IDIM - 32; k0 += 32) {
    v16bf a0  = load_a_frag(sA, 0,  k0, lane);
    v16bf a1  = load_a_frag(sA, 16, k0, lane);
    v16bf nb0 = load_b_frag(Wb, IDIM, n0,      k0 + 32, lane);
    v16bf nb1 = load_b_frag(Wb, IDIM, n0 + 16, k0 + 32, lane);
    if (k0 + 256 < IDIM) {
      const int n = lane & 15, kb = k0 + 256 + ((lane >> 4) << 4);
      __builtin_prefetch(Wb + (size_t)(n0 + n) * IDIM + kb, 0, 3);
    }
    c00 = __builtin_amdgcn_wmma_f32_16x16x32_bf16(false, a0, false, b0, (short)0, c00, false, false);
    c10 = __builtin_amdgcn_wmma_f32_16x16x32_bf16(false, a1, false, b0, (short)0, c10, false, false);
    c01 = __builtin_amdgcn_wmma_f32_16x16x32_bf16(false, a0, false, b1, (short)0, c01, false, false);
    c11 = __builtin_amdgcn_wmma_f32_16x16x32_bf16(false, a1, false, b1, (short)0, c11, false, false);
    b0 = nb0; b1 = nb1;
  }
  {
    const int k0 = IDIM - 32;
    v16bf a0 = load_a_frag(sA, 0,  k0, lane);
    v16bf a1 = load_a_frag(sA, 16, k0, lane);
    c00 = __builtin_amdgcn_wmma_f32_16x16x32_bf16(false, a0, false, b0, (short)0, c00, false, false);
    c10 = __builtin_amdgcn_wmma_f32_16x16x32_bf16(false, a1, false, b0, (short)0, c10, false, false);
    c01 = __builtin_amdgcn_wmma_f32_16x16x32_bf16(false, a0, false, b1, (short)0, c01, false, false);
    c11 = __builtin_amdgcn_wmma_f32_16x16x32_bf16(false, a1, false, b1, (short)0, c11, false, false);
  }

  const int n  = lane & 15;
  const int mh = (lane >> 4) << 3;
#pragma unroll
  for (int nh = 0; nh < 2; ++nh) {
    const int hc = n0 + nh * 16 + n;
    const float bias = __bfloat162float(mlp2_b[(size_t)e * HDIM + hc]);
#pragma unroll
    for (int half = 0; half < 2; ++half) {
      v8f c = nh ? (half ? c11 : c01) : (half ? c10 : c00);
#pragma unroll
      for (int r = 0; r < 8; ++r) {
        const int m = half * 16 + r + mh;
        if (m0 + m < cnt) {
          const int slot = seg + m0 + m;
          down_buf[(size_t)slot * HDIM + hc] = slot_w[slot] * (c[r] + bias);
        }
      }
    }
  }
}

// ---------------------------------------------------------------------------
__global__ void k_combine(const __hip_bfloat16* __restrict__ x,
                          const float* __restrict__ down_buf,
                          const int*   __restrict__ inv,
                          __hip_bfloat16* __restrict__ out) {
  const int i = blockIdx.x * blockDim.x + threadIdx.x;
  const int t = i >> 10;
  const int h = i & (HDIM - 1);
  float v = __bfloat162float(x[i]);
  v += down_buf[(size_t)inv[t * 2 + 0] * HDIM + h];
  v += down_buf[(size_t)inv[t * 2 + 1] * HDIM + h];
  out[i] = __float2bfloat16(v);
}

// ---------------------------------------------------------------------------
extern "C" void kernel_launch(void* const* d_in, const int* in_sizes, int n_in,
                              void* d_out, int out_size, void* d_ws, size_t ws_size,
                              hipStream_t stream) {
  const __hip_bfloat16* x      = (const __hip_bfloat16*)d_in[0];
  const float*          nsc    = (const float*)         d_in[1];
  const __hip_bfloat16* gate_w = (const __hip_bfloat16*)d_in[2];
  const __hip_bfloat16* gate_b = (const __hip_bfloat16*)d_in[3];
  const __hip_bfloat16* mlp1_w = (const __hip_bfloat16*)d_in[4];
  const __hip_bfloat16* mlp1_b = (const __hip_bfloat16*)d_in[5];
  const __hip_bfloat16* mlp2_w = (const __hip_bfloat16*)d_in[6];
  const __hip_bfloat16* mlp2_b = (const __hip_bfloat16*)d_in[7];
  __hip_bfloat16*       out    = (__hip_bfloat16*)d_out;

  char* ws = (char*)d_ws;
  size_t off = 0;
  auto alloc = [&](size_t bytes) { void* p = ws + off; off = (off + bytes + 255) & ~(size_t)255; return p; };
  __hip_bfloat16* t_norm  = (__hip_bfloat16*)alloc((size_t)T_TOK * HDIM * 2);
  __hip_bfloat16* h_act   = (__hip_bfloat16*)alloc((size_t)T_TOK * TOPK * IDIM * 2);
  float*          downbuf = (float*)         alloc((size_t)T_TOK * TOPK * HDIM * 4);
  int*            counts  = (int*)           alloc(NEXP * 4);
  int*            offs    = (int*)           alloc(NEXP * 4);
  int*            cursor  = (int*)           alloc(NEXP * 4);
  int*            perm    = (int*)           alloc((size_t)T_TOK * TOPK * 4);
  int*            inv     = (int*)           alloc((size_t)T_TOK * TOPK * 4);
  float*          slot_w  = (float*)         alloc((size_t)T_TOK * TOPK * 4);
  int*            idx_tk  = (int*)           alloc((size_t)T_TOK * TOPK * 4);
  float*          ew_tk   = (float*)         alloc((size_t)T_TOK * TOPK * 4);

  k_zero<<<1, 32, 0, stream>>>(counts);
  k_route<<<T_TOK, 128, 0, stream>>>(x, nsc, gate_w, gate_b, t_norm, idx_tk, ew_tk, counts);
  k_prefix<<<1, 1, 0, stream>>>(counts, offs, cursor);
  k_scatter<<<(T_TOK + 255) / 256, 256, 0, stream>>>(idx_tk, ew_tk, cursor, perm, inv, slot_w);
  k_up<<<dim3(TWO_I / 256, T_TOK / 32, NEXP), 256, 0, stream>>>(
      t_norm, mlp1_w, mlp1_b, perm, offs, counts, h_act);
  k_down<<<dim3(HDIM / 256, T_TOK / 32, NEXP), 256, 0, stream>>>(
      h_act, mlp2_w, mlp2_b, offs, counts, slot_w, downbuf);
  k_combine<<<(T_TOK * HDIM) / 256, 256, 0, stream>>>(x, downbuf, inv, out);
}
